// Attention_32049045963351
// MI455X (gfx1250) — matrix operand order repaired and checked by reading, not compiled, not run on hardware
//
#include <hip/hip_runtime.h>
#include <hip/hip_bf16.h>

namespace {

constexpr int S_LEN = 2048;
constexpr int DIM   = 4096;
constexpr int NHEAD = 32;
constexpr int NKVH  = 8;
constexpr int HDIM  = 128;

typedef __attribute__((ext_vector_type(16))) __bf16        v16bf;
typedef __attribute__((ext_vector_type(8)))  float          v8f;
typedef __attribute__((ext_vector_type(8)))  unsigned short v8u;
typedef __attribute__((ext_vector_type(16))) unsigned short v16u;

__device__ __forceinline__ unsigned short f2bf(float f) {
  unsigned int u = __float_as_uint(f);
  u += 0x7FFFu + ((u >> 16) & 1u);   // round-to-nearest-even
  return (unsigned short)(u >> 16);
}

__device__ __forceinline__ v16bf make_frag(v8u lo, v8u hi) {
  v16u t;
#pragma unroll
  for (int i = 0; i < 8; ++i) { t[i] = lo[i]; t[i + 8] = hi[i]; }
  return __builtin_bit_cast(v16bf, t);
}

// A fragment (16x32, MxK) from row-major source with leading dim ld (elements).
// ISA layout: lane<16 -> M=lane, K = {0..7, 16..23}; lane>=16 -> M=lane-16, K = {8..15, 24..31}
__device__ __forceinline__ v16bf load_frag_a(const unsigned short* base, int ld, int m0, int k0) {
  const int lane = threadIdx.x & 31;
  const int hlf  = lane >> 4;
  const unsigned short* p = base + (size_t)(m0 + (lane & 15)) * ld + k0 + hlf * 8;
  v8u lo = *(const v8u*)(p);        // K = 8h .. 8h+7
  v8u hi = *(const v8u*)(p + 16);   // K = 16+8h .. 16+8h+7
  return make_frag(lo, hi);
}

// B fragment (32x16, KxN) from a row-major [N,K] source (ld in elements).
// ISA layout: lane<16 -> N=lane, K = {0..7, 16..23}; lane>=16 -> N=lane-16, K = {8..15, 24..31}
__device__ __forceinline__ v16bf load_frag_b(const unsigned short* base, int ld, int n0, int k0) {
  const int lane = threadIdx.x & 31;
  const unsigned short* p = base + (size_t)(n0 + (lane & 15)) * ld + k0 + (lane >> 4) * 8;
  v8u lo = *(const v8u*)(p);
  v8u hi = *(const v8u*)(p + 16);
  return make_frag(lo, hi);
}

__device__ __forceinline__ v8f wmma_bf16(v16bf a, v16bf b, v8f c) {
  return __builtin_amdgcn_wmma_f32_16x16x32_bf16(false, a, false, b, (short)0, c, false, false);
}

// Async global->LDS copy of 16 bytes (GLOBAL_LOAD_ASYNC_TO_LDS_B128, ASYNCcnt).
__device__ __forceinline__ void async_ld_b128(unsigned lds_addr, unsigned long long gaddr) {
  asm volatile("global_load_async_to_lds_b128 %0, %1, off"
               :: "v"(lds_addr), "v"(gaddr) : "memory");
}
__device__ __forceinline__ void wait_asynccnt0() {
  asm volatile("s_wait_asynccnt 0" ::: "memory");
}

// ---------------------------------------------------------------- conversions

__global__ void cvt_f32_bf16(const float* __restrict__ in, unsigned short* __restrict__ out, int n) {
  int i = blockIdx.x * blockDim.x + threadIdx.x;
  if (i < n) out[i] = f2bf(in[i]);
}

// RoPE on (S, nheads, HDIM) f32, write bf16 (scale folded in; scale=1/sqrt(HD) for Q).
__global__ void rope_cvt(const float* __restrict__ in, const float* __restrict__ fr,
                         unsigned short* __restrict__ out, int nheads, float scale) {
  int idx = blockIdx.x * blockDim.x + threadIdx.x;
  int total = S_LEN * nheads * (HDIM / 2);
  if (idx >= total) return;
  int i = idx % (HDIM / 2);
  int h = (idx / (HDIM / 2)) % nheads;
  int s = idx / ((HDIM / 2) * nheads);
  float c  = fr[s * HDIM + 2 * i];
  float sn = fr[s * HDIM + 2 * i + 1];
  size_t off = (size_t)s * nheads * HDIM + (size_t)h * HDIM + 2 * i;
  float x0 = in[off], x1 = in[off + 1];
  out[off]     = f2bf((x0 * c - x1 * sn) * scale);
  out[off + 1] = f2bf((x0 * sn + x1 * c) * scale);
}

// V (S, NKVH*HDIM) f32 -> Vt (NKVH*HDIM, S) bf16  (so P@V B-fragments are contiguous)
__global__ void vtrans_cvt(const float* __restrict__ in, unsigned short* __restrict__ out) {
  int idx = blockIdx.x * blockDim.x + threadIdx.x;
  int s = idx % S_LEN;
  int c = idx / S_LEN;
  if (c >= NKVH * HDIM) return;
  out[(size_t)c * S_LEN + s] = f2bf(in[(size_t)s * (NKVH * HDIM) + c]);
}

// ---------------------------------------------------------------- GEMM C = A(MxK) * B(NxK)^T
// 256 threads = 8 waves; block tile 64x256; wave tile 32x64 (2x4 WMMA tiles).
// Operands are L2-resident (192MB L2); direct global fragment loads pipeline well.
__global__ __launch_bounds__(256) void gemm_nt_bf16(
    const unsigned short* __restrict__ A, const unsigned short* __restrict__ B,
    float* __restrict__ C, int M, int N, int K) {
  const int wave = threadIdx.x >> 5;
  const int m0 = blockIdx.y * 64  + (wave >> 2) * 32;
  const int n0 = blockIdx.x * 256 + (wave & 3) * 64;

  v8f acc[2][4];
#pragma unroll
  for (int i = 0; i < 2; ++i)
#pragma unroll
    for (int j = 0; j < 4; ++j) acc[i][j] = {};

  for (int k0 = 0; k0 < K; k0 += 32) {
    v16bf a0 = load_frag_a(A, K, m0,      k0);
    v16bf a1 = load_frag_a(A, K, m0 + 16, k0);
#pragma unroll
    for (int j = 0; j < 4; ++j) {
      v16bf b = load_frag_b(B, K, n0 + 16 * j, k0);
      acc[0][j] = wmma_bf16(a0, b, acc[0][j]);
      acc[1][j] = wmma_bf16(a1, b, acc[1][j]);
    }
  }

  const int lane = threadIdx.x & 31;
  const int col  = lane & 15;
  const int rb   = 8 * (lane >> 4);
#pragma unroll
  for (int i = 0; i < 2; ++i)
#pragma unroll
    for (int j = 0; j < 4; ++j)
#pragma unroll
      for (int r = 0; r < 8; ++r)
        C[(size_t)(m0 + 16 * i + rb + r) * N + (n0 + 16 * j + col)] = acc[i][j][r];
}

// ---------------------------------------------------------------- flash attention
// 4 waves / block, 64 query rows of one head. K/V tiles for each 32-key step are
// staged into LDS with async global->LDS DMA and shared by all 4 waves (GQA reuse).
__global__ __launch_bounds__(128) void attn_fa(const unsigned short* __restrict__ Qb,
                                               const unsigned short* __restrict__ Kb,
                                               const unsigned short* __restrict__ Vt,
                                               float* __restrict__ Y) {
  __shared__ unsigned short kt[32 * 128];        // K tile: 32 keys x 128 dims (8KB)
  __shared__ unsigned short vt[128 * 32];        // V tile: 128 dims x 32 keys (8KB)
  __shared__ unsigned short pb[4][16 * 32];      // per-wave P staging (4KB)

  const int tid  = threadIdx.x;
  const int lane = tid & 31;
  const int wid  = tid >> 5;
  const int hlf  = lane >> 4;
  const int h    = blockIdx.y;
  const int sb   = blockIdx.x * 64;
  const int s0   = sb + wid * 16;
  const int kvh  = h >> 2;                       // REPEATS = NH/NKV = 4
  const int blk_smax = sb + 63;
  const int my_smax  = s0 + 15;

  const unsigned kt_base = (unsigned)(uintptr_t)&kt[0];
  const unsigned vt_base = (unsigned)(uintptr_t)&vt[0];
  const unsigned long long kb_addr = (unsigned long long)(uintptr_t)Kb;
  const unsigned long long vt_addr = (unsigned long long)(uintptr_t)Vt;

  // Per-wave Q fragments: 16x128 (4 x K=32), scale already folded in.
  v16bf qf[4];
#pragma unroll
  for (int kk = 0; kk < 4; ++kk)
    qf[kk] = load_frag_a(Qb, NHEAD * HDIM, s0, h * HDIM + kk * 32);

  v8f o[8];
#pragma unroll
  for (int i = 0; i < 8; ++i) o[i] = {};
  float m[8], l[8];
#pragma unroll
  for (int r = 0; r < 8; ++r) { m[r] = -3.0e38f; l[r] = 0.0f; }

  for (int t0 = 0; t0 <= blk_smax; t0 += 32) {
    __syncthreads();   // all waves done reading previous K/V tiles

    // ---- cooperative async stage of K tile (32 rows x 256B) and V tile (128 rows x 64B)
    {
      const unsigned long long kg = kb_addr + ((size_t)t0 * (NKVH * HDIM) + (size_t)kvh * HDIM) * 2;
#pragma unroll
      for (int i = 0; i < 4; ++i) {
        int c = tid + 128 * i;                 // 512 chunks of 16B
        int row = c >> 4, colb = (c & 15) * 16;
        async_ld_b128(kt_base + row * 256 + colb,
                      kg + (size_t)row * (NKVH * HDIM * 2) + colb);
      }
      const unsigned long long vg = vt_addr + ((size_t)kvh * HDIM * S_LEN + (size_t)t0) * 2;
#pragma unroll
      for (int i = 0; i < 4; ++i) {
        int c = tid + 128 * i;                 // 512 chunks of 16B
        int row = c >> 2, colb = (c & 3) * 16;
        async_ld_b128(vt_base + row * 64 + colb,
                      vg + (size_t)row * (S_LEN * 2) + colb);
      }
      wait_asynccnt0();
    }
    __syncthreads();   // tiles visible to all waves

    if (t0 <= my_smax) {
      // S = Q K^T for 32 keys (two 16x16 C tiles), K fragments from LDS
      v8f st[2];
#pragma unroll
      for (int j = 0; j < 2; ++j) {
        v8f c = {};
#pragma unroll
        for (int kk = 0; kk < 4; ++kk) {
          v16bf kf = load_frag_b(kt, HDIM, 16 * j, kk * 32);
          c = wmma_bf16(qf[kk], kf, c);
        }
        st[j] = c;
      }
      // causal mask
#pragma unroll
      for (int j = 0; j < 2; ++j) {
        int t = t0 + 16 * j + (lane & 15);
#pragma unroll
        for (int r = 0; r < 8; ++r)
          if (t > s0 + r + 8 * hlf) st[j][r] = -3.0e38f;
      }
      // online softmax: row reductions across the 16 lanes of each half
      float rmax[8];
#pragma unroll
      for (int r = 0; r < 8; ++r) rmax[r] = fmaxf(st[0][r], st[1][r]);
#pragma unroll
      for (int msk = 1; msk < 16; msk <<= 1)
#pragma unroll
        for (int r = 0; r < 8; ++r) rmax[r] = fmaxf(rmax[r], __shfl_xor(rmax[r], msk, 32));

      float alpha[8];
#pragma unroll
      for (int r = 0; r < 8; ++r) {
        float mn = fmaxf(m[r], rmax[r]);
        alpha[r] = __expf(m[r] - mn);
        m[r] = mn;
      }
      float rsum[8];
#pragma unroll
      for (int r = 0; r < 8; ++r) rsum[r] = 0.0f;
#pragma unroll
      for (int j = 0; j < 2; ++j)
#pragma unroll
        for (int r = 0; r < 8; ++r) {
          float p = __expf(st[j][r] - m[r]);
          st[j][r] = p;
          rsum[r] += p;
        }
#pragma unroll
      for (int msk = 1; msk < 16; msk <<= 1)
#pragma unroll
        for (int r = 0; r < 8; ++r) rsum[r] += __shfl_xor(rsum[r], msk, 32);
#pragma unroll
      for (int r = 0; r < 8; ++r) l[r] = l[r] * alpha[r] + rsum[r];
#pragma unroll
      for (int i = 0; i < 8; ++i)
#pragma unroll
        for (int r = 0; r < 8; ++r) o[i][r] *= alpha[r];

      // P (16x32) through this wave's private LDS region: C-layout -> A-layout.
      // Same-wave LDS ops are executed in order, no barrier needed.
      unsigned short* pw = &pb[wid][0];
#pragma unroll
      for (int j = 0; j < 2; ++j)
#pragma unroll
        for (int r = 0; r < 8; ++r)
          pw[(r + 8 * hlf) * 32 + 16 * j + (lane & 15)] = f2bf(st[j][r]);

      v16bf pf;
      {
        int row = lane & 15;
        v8u lo = *(const v8u*)&pw[row * 32 + 8 * hlf];
        v8u hi = *(const v8u*)&pw[row * 32 + 16 + 8 * hlf];
        pf = make_frag(lo, hi);
      }
      // O += P @ V, V fragments from LDS (vt tile: [d][32 keys])
#pragma unroll
      for (int dt = 0; dt < 8; ++dt) {
        v16bf vf = load_frag_b(vt, 32, dt * 16, 0);
        o[dt] = wmma_bf16(pf, vf, o[dt]);
      }
    }
  }

  // normalize + store
#pragma unroll
  for (int dt = 0; dt < 8; ++dt)
#pragma unroll
    for (int r = 0; r < 8; ++r)
      Y[(size_t)(s0 + r + 8 * hlf) * (NHEAD * HDIM) + h * HDIM + dt * 16 + (lane & 15)] =
          o[dt][r] * (1.0f / l[r]);
}

}  // namespace

extern "C" void kernel_launch(void* const* d_in, const int* in_sizes, int n_in,
                              void* d_out, int out_size, void* d_ws, size_t ws_size,
                              hipStream_t stream) {
  (void)in_sizes; (void)n_in; (void)out_size; (void)ws_size;
  const float* x  = (const float*)d_in[0];
  const float* fr = (const float*)d_in[1];
  const float* wq = (const float*)d_in[2];
  const float* wk = (const float*)d_in[3];
  const float* wv = (const float*)d_in[4];
  const float* wo = (const float*)d_in[5];
  float* out = (float*)d_out;

  const size_t nX  = (size_t)S_LEN * DIM;            // 8M elems
  const size_t nWQ = (size_t)DIM * DIM;              // 16M
  const size_t nWK = (size_t)(NKVH * HDIM) * DIM;    // 4M
  const size_t nKV = (size_t)S_LEN * NKVH * HDIM;    // 2M

  char* p = (char*)d_ws;
  auto take = [&](size_t bytes) { char* q = p; p += bytes; return q; };
  unsigned short* xb  = (unsigned short*)take(nX  * 2);
  unsigned short* wqb = (unsigned short*)take(nWQ * 2);
  unsigned short* wkb = (unsigned short*)take(nWK * 2);
  unsigned short* wvb = (unsigned short*)take(nWK * 2);
  unsigned short* wob = (unsigned short*)take(nWQ * 2);
  float* Qf = (float*)take(nX  * 4);   // reused as attention output Yf
  float* Kf = (float*)take(nKV * 4);   // Kf+Vf region (16 MiB) reused as Yb
  float* Vf = (float*)take(nKV * 4);
  unsigned short* Qb = (unsigned short*)take(nX  * 2);
  unsigned short* Kb = (unsigned short*)take(nKV * 2);
  unsigned short* Vt = (unsigned short*)take(nKV * 2);
  float* Yf = Qf;
  unsigned short* Yb = (unsigned short*)Kf;

  // 1) f32 -> bf16
  cvt_f32_bf16<<<(int)((nX  + 255) / 256), 256, 0, stream>>>(x,  xb,  (int)nX);
  cvt_f32_bf16<<<(int)((nWQ + 255) / 256), 256, 0, stream>>>(wq, wqb, (int)nWQ);
  cvt_f32_bf16<<<(int)((nWK + 255) / 256), 256, 0, stream>>>(wk, wkb, (int)nWK);
  cvt_f32_bf16<<<(int)((nWK + 255) / 256), 256, 0, stream>>>(wv, wvb, (int)nWK);
  cvt_f32_bf16<<<(int)((nWQ + 255) / 256), 256, 0, stream>>>(wo, wob, (int)nWQ);

  // 2) Q/K/V projections (WMMA GEMMs)
  gemm_nt_bf16<<<dim3(DIM / 256, S_LEN / 64), 256, 0, stream>>>(xb, wqb, Qf, S_LEN, DIM, DIM);
  gemm_nt_bf16<<<dim3((NKVH * HDIM) / 256, S_LEN / 64), 256, 0, stream>>>(xb, wkb, Kf, S_LEN, NKVH * HDIM, DIM);
  gemm_nt_bf16<<<dim3((NKVH * HDIM) / 256, S_LEN / 64), 256, 0, stream>>>(xb, wvb, Vf, S_LEN, NKVH * HDIM, DIM);

  // 3) RoPE (1/sqrt(HD) folded into Q) + V transpose, all to bf16
  const float scale = 0.08838834764831845f;  // 1/sqrt(128)
  int nq = S_LEN * NHEAD * (HDIM / 2);
  rope_cvt<<<(nq + 255) / 256, 256, 0, stream>>>(Qf, fr, Qb, NHEAD, scale);
  int nk = S_LEN * NKVH * (HDIM / 2);
  rope_cvt<<<(nk + 255) / 256, 256, 0, stream>>>(Kf, fr, Kb, NKVH, 1.0f);
  int nv = NKVH * HDIM * S_LEN;
  vtrans_cvt<<<(nv + 255) / 256, 256, 0, stream>>>(Vf, Vt);

  // 4) causal GQA flash attention (WMMA + async global->LDS K/V staging)
  attn_fa<<<dim3(S_LEN / 64, NHEAD), 128, 0, stream>>>(Qb, Kb, Vt, Yf);

  // 5) output projection
  cvt_f32_bf16<<<(int)((nX + 255) / 256), 256, 0, stream>>>(Yf, Yb, (int)nX);
  gemm_nt_bf16<<<dim3(DIM / 256, S_LEN / 64), 256, 0, stream>>>(Yb, wob, out, S_LEN, DIM, DIM);
}
